// MRFLoss_31542239822308
// MI455X (gfx1250) — compile-verified, matrix operand order checked by reading, not run on hardware
//
#include <hip/hip_runtime.h>

typedef _Float16 v8h  __attribute__((ext_vector_type(8)));
typedef _Float16 v16h __attribute__((ext_vector_type(16)));
typedef float    v8f  __attribute__((ext_vector_type(8)));

#define HCN   256                 // channels
#define HW    64                  // spatial size
#define OW    62                  // patches per row (64-3+1)
#define NP    3844                // 62*62 patches
#define KTOT  2304                // 256*9
#define KC    (KTOT / 32)         // 72 K-steps of 32
#define TQ    64                  // q rows per block
#define TP    128                 // p cols per block
#define NQT   61                  // ceil(3844/64)
#define NPT   31                  // ceil(3844/128)
#define XNB   (4ull*HW*HW*HCN*2)  // bytes of one normalized f16 field (8 MB)

__device__ __forceinline__ unsigned int f2ord(float f) {
    unsigned int u = __float_as_uint(f);
    return (u & 0x80000000u) ? ~u : (u | 0x80000000u);
}

// CDNA5 async global->LDS copy (ASYNCcnt-tracked), GV addressing mode.
__device__ __forceinline__ void async_ld_b128(unsigned lds_byte_off, const void* gptr) {
    asm volatile("global_load_async_to_lds_b128 %0, %1, off"
                 :: "v"(lds_byte_off), "v"((unsigned long long)(uintptr_t)gptr)
                 : "memory");
}
__device__ __forceinline__ void wait_async0() {
    asm volatile("s_wait_asynccnt 0x0" ::: "memory");
}
__device__ __forceinline__ unsigned lds_off(const void* p) {
    return (unsigned)(uintptr_t)p;   // low 32 bits of generic LDS ptr = LDS offset
}

// ---------------------------------------------------------------------------
// Kernel 1: per-pixel channel L2-normalize (B,C,H,W) f32 -> (B,H,W,C) f16
// ---------------------------------------------------------------------------
__global__ __launch_bounds__(256) void mrf_norm(const float* __restrict__ in,
                                                _Float16* __restrict__ out) {
    const int y = blockIdx.x, b = blockIdx.y, t = threadIdx.x;
    __shared__ float red[4][64];
    __shared__ float sinv[64];
    {
        const int x = t & 63, cg = t >> 6;
        const float* base = in + ((size_t)b * HCN * HW + y) * HW + x;
        float ss = 0.f;
        for (int c = cg; c < HCN; c += 4) {
            float v = base[(size_t)c * HW * HW];
            ss += v * v;
        }
        red[cg][x] = ss;
    }
    __syncthreads();
    if (t < 64) {
        float n = sqrtf(red[0][t] + red[1][t] + red[2][t] + red[3][t]);
        sinv[t] = 1.0f / fmaxf(n, 1e-12f);
    }
    __syncthreads();
    {
        const int x = t >> 2, cs = (t & 3) * 64;
        const float inv = sinv[x];
        const float* base = in + ((size_t)b * HCN * HW + y) * HW + x;
        _Float16* obase = out + (((size_t)b * HW + y) * HW + x) * HCN + cs;
        for (int cc = 0; cc < 64; cc += 8) {
            v8h h;
#pragma unroll
            for (int u = 0; u < 8; ++u)
                h[u] = (_Float16)(base[(size_t)(cs + cc + u) * HW * HW] * inv);
            *(v8h*)(obase + cc) = h;
        }
    }
}

// ---------------------------------------------------------------------------
// Kernel 2: fused 3844x3844 (K=2304) f16 WMMA GEMM + row argmax
// block tile 64q x 128p, 8 waves (4x2), wave = 16x64 strip = 4 accumulators.
// Double-buffered async global->LDS pipeline (ASYNCcnt).
// ---------------------------------------------------------------------------
__global__ __launch_bounds__(256) void mrf_match(const _Float16* __restrict__ xn,
                                                 const _Float16* __restrict__ tn,
                                                 int* __restrict__ match) {
    const int b = blockIdx.y;
    const int qbase = blockIdx.x * TQ;
    const int t = threadIdx.x;
    const int lane = t & 31, wave = t >> 5;
    const int wr = wave >> 1, wc = wave & 1;

    __shared__ __align__(16) _Float16 At[2][TQ][40];   // 2 x 5120 B
    __shared__ __align__(16) _Float16 Bt[2][TP][40];   // 2 x 10240 B
    __shared__ unsigned long long best[TQ];

    if (t < TQ) best[t] = 0ull;                        // < f2ord(any finite float)

    // ---- loader mapping: A: 1 x 16B per thread, B: 2 x 16B per thread ----
    const int lrow = t >> 2, lseg = t & 3;             // row 0..63, 8-half segment
    const int q  = qbase + lrow;
    const bool qok = q < NP;
    const int qy = qok ? q / OW : 0, qx = qok ? q - qy * OW : 0;
    const _Float16* xb = xn + (size_t)b * HW * HW * HCN;
    const _Float16* tb = tn + (size_t)b * HW * HW * HCN;
    const _Float16* abase = xb + ((size_t)(qy * HW + qx)) * HCN + lseg * 8;

    const unsigned aL[2] = { lds_off(&At[0][lrow][lseg * 8]),
                             lds_off(&At[1][lrow][lseg * 8]) };
    const unsigned b0L[2] = { lds_off(&Bt[0][lrow][lseg * 8]),
                              lds_off(&Bt[1][lrow][lseg * 8]) };
    const unsigned b1L[2] = { lds_off(&Bt[0][64 + lrow][lseg * 8]),
                              lds_off(&Bt[1][64 + lrow][lseg * 8]) };

    // A rows beyond NP: zero once in both buffers; lanes never overwrite them.
    if (!qok) {
        *(uint4*)&At[0][lrow][lseg * 8] = make_uint4(0u, 0u, 0u, 0u);
        *(uint4*)&At[1][lrow][lseg * 8] = make_uint4(0u, 0u, 0u, 0u);
    }

    // ---- WMMA fragment mapping (ISA 7.12.2, 16-bit A 16x32) ----
    const int base0 = (lane < 16) ? 0 : 8;
    const int arow  = wr * 16 + (lane & 15);
    const int brow  = wc * 64 + (lane & 15);

    for (int pt = 0; pt < NPT; ++pt) {
        const int pbase = pt * TP;
        const int p0 = pbase + lrow, p1 = p0 + 64;
        const bool p0ok = p0 < NP, p1ok = p1 < NP;
        const int p0y = p0ok ? p0 / OW : 0, p0x = p0ok ? p0 - p0y * OW : 0;
        const int p1y = p1ok ? p1 / OW : 0, p1x = p1ok ? p1 - p1y * OW : 0;
        const _Float16* bbase0 = tb + ((size_t)(p0y * HW + p0x)) * HCN + lseg * 8;
        const _Float16* bbase1 = tb + ((size_t)(p1y * HW + p1x)) * HCN + lseg * 8;

        __syncthreads();   // previous tile's reads done before refill/zero

        if (!p0ok) {
            *(uint4*)&Bt[0][lrow][lseg * 8] = make_uint4(0u, 0u, 0u, 0u);
            *(uint4*)&Bt[1][lrow][lseg * 8] = make_uint4(0u, 0u, 0u, 0u);
        }
        if (!p1ok) {
            *(uint4*)&Bt[0][64 + lrow][lseg * 8] = make_uint4(0u, 0u, 0u, 0u);
            *(uint4*)&Bt[1][64 + lrow][lseg * 8] = make_uint4(0u, 0u, 0u, 0u);
        }

        // issue stage 0 into buffer 0
        {
            if (qok)  async_ld_b128(aL[0],  abase);
            if (p0ok) async_ld_b128(b0L[0], bbase0);
            if (p1ok) async_ld_b128(b1L[0], bbase1);
        }

        v8f acc0 = {}, acc1 = {}, acc2 = {}, acc3 = {};

        for (int kc = 0; kc < KC; ++kc) {
            const int cur = kc & 1;
            wait_async0();      // stage kc landed (this wave)
            __syncthreads();    // landed everywhere; prev reads of ~cur done

            if (kc + 1 < KC) {  // issue stage kc+1 into ~cur, overlaps compute
                const int s = kc + 1;
                const int ij = s >> 3;
                const int di = ij / 3, dj = ij - di * 3;
                const int off = (di * HW + dj) * HCN + (s & 7) * 32;  // halves
                if (qok)  async_ld_b128(aL[cur ^ 1],  abase  + off);
                if (p0ok) async_ld_b128(b0L[cur ^ 1], bbase0 + off);
                if (p1ok) async_ld_b128(b1L[cur ^ 1], bbase1 + off);
            }

            // Load ALL fragments, then the four WMMAs; sched_group_barrier
            // pins the order (10 DS-reads, then 4 WMMAs) so all fragments
            // stay live in distinct registers and the WMMAs issue back-to-back
            // behind a single dscnt wait.
            v16h a, bf0, bf1, bf2, bf3;
            {
                const v8h lo = *(const v8h*)&At[cur][arow][base0];
                const v8h hi = *(const v8h*)&At[cur][arow][base0 + 16];
#pragma unroll
                for (int u = 0; u < 8; ++u) { a[u] = lo[u]; a[u + 8] = hi[u]; }
            }
#define BFRAG(bv, roff)                                                     \
            {                                                               \
                const v8h lo = *(const v8h*)&Bt[cur][brow + (roff)][base0]; \
                const v8h hi = *(const v8h*)&Bt[cur][brow + (roff)][base0 + 16]; \
                _Pragma("unroll")                                           \
                for (int u = 0; u < 8; ++u) { bv[u] = lo[u]; bv[u + 8] = hi[u]; } \
            }
            BFRAG(bf0, 0)
            BFRAG(bf1, 16)
            BFRAG(bf2, 32)
            BFRAG(bf3, 48)
            acc0 = __builtin_amdgcn_wmma_f32_16x16x32_f16(false, a, false, bf0,
                                                          (short)0, acc0, false, false);
            acc1 = __builtin_amdgcn_wmma_f32_16x16x32_f16(false, a, false, bf1,
                                                          (short)0, acc1, false, false);
            acc2 = __builtin_amdgcn_wmma_f32_16x16x32_f16(false, a, false, bf2,
                                                          (short)0, acc2, false, false);
            acc3 = __builtin_amdgcn_wmma_f32_16x16x32_f16(false, a, false, bf3,
                                                          (short)0, acc3, false, false);
            // scheduling groups: 0x100 = DS read, 0x008 = MFMA/WMMA
            __builtin_amdgcn_sched_group_barrier(0x100, 10, 0);
            __builtin_amdgcn_sched_group_barrier(0x008, 4, 0);
#undef BFRAG
        }

        // ---- per-row argmax over this 64x128 tile ----
        // D layout: VGPR r -> row r (lanes 0-15) / row 8+r (lanes 16-31); col = lane%16
#pragma unroll
        for (int r = 0; r < 8; ++r) {
            const int m = wr * 16 + ((lane < 16) ? r : (8 + r));
            const float vv[4] = { acc0[r], acc1[r], acc2[r], acc3[r] };
#pragma unroll
            for (int tt = 0; tt < 4; ++tt) {
                const int col = pbase + wc * 64 + tt * 16 + (lane & 15);
                unsigned long long pk = (col < NP)
                    ? ((((unsigned long long)f2ord(vv[tt])) << 32) |
                       (unsigned long long)(4095 - col))
                    : 0ull;
#pragma unroll
                for (int off = 1; off < 16; off <<= 1) {
                    unsigned long long o = __shfl_xor(pk, off, 32);
                    if (o > pk) pk = o;
                }
                if ((lane & 15) == 0) atomicMax(&best[m], pk);
            }
        }
    }
    __syncthreads();
    if (t < TQ && (qbase + t) < NP)
        match[(size_t)b * NP + qbase + t] = 4095 - (int)(best[t] & 0xFFFFFFFFu);
}

// ---------------------------------------------------------------------------
// Kernel 3: MSE over (pred patch q) vs (target patch match[q]), f32 data
// ---------------------------------------------------------------------------
__global__ __launch_bounds__(256) void mrf_mse(const float* __restrict__ pred,
                                               const float* __restrict__ targ,
                                               const int* __restrict__ match,
                                               float* __restrict__ out) {
    const int q = blockIdx.x, b = blockIdx.y, t = threadIdx.x;
    const int m = match[(size_t)b * NP + q];
    const int qy = q / OW, qx = q - qy * OW;
    const int py = m / OW, px = m - py * OW;
    const float* pb = pred + (size_t)b * HCN * HW * HW;
    const float* tp = targ + (size_t)b * HCN * HW * HW;
    float s = 0.f;
    for (int idx = t; idx < KTOT; idx += 256) {
        const int ij = idx >> 8, c = idx & 255;
        const int di = ij / 3, dj = ij - di * 3;
        float pv = pb[((size_t)c * HW + qy + di) * HW + qx + dj];
        float tv = tp[((size_t)c * HW + py + di) * HW + px + dj];
        float d = pv - tv;
        s += d * d;
    }
    __shared__ float red[256];
    red[t] = s;
    __syncthreads();
    for (int off = 128; off > 0; off >>= 1) {
        if (t < off) red[t] += red[t + off];
        __syncthreads();
    }
    if (t == 0)
        atomicAdd(out, red[0] * (1.0f / ((float)4 * NP * KTOT)));
}

// ---------------------------------------------------------------------------
extern "C" void kernel_launch(void* const* d_in, const int* in_sizes, int n_in,
                              void* d_out, int out_size, void* d_ws, size_t ws_size,
                              hipStream_t stream) {
    const float* pred = (const float*)d_in[0];
    const float* targ = (const float*)d_in[1];
    char* ws = (char*)d_ws;
    _Float16* xn = (_Float16*)ws;             // 8 MB normalized pred (B,H,W,C) f16
    _Float16* tn = (_Float16*)(ws + XNB);     // 8 MB normalized target
    int* match = (int*)(ws + 2 * XNB);        // 4*3844 ints

    hipMemsetAsync(d_out, 0, sizeof(float), stream);

    dim3 gN(HW, 4);
    mrf_norm<<<gN, 256, 0, stream>>>(pred, xn);
    mrf_norm<<<gN, 256, 0, stream>>>(targ, tn);

    dim3 gG(NQT, 4);
    mrf_match<<<gG, 256, 0, stream>>>(xn, tn, match);

    dim3 gM(NP, 4);
    mrf_mse<<<gM, 256, 0, stream>>>(pred, targ, match, (float*)d_out);
}
